// MultiHeadAttention_67808943669749
// MI455X (gfx1250) — matrix operand order repaired and checked by reading, not compiled, not run on hardware
//
#include <hip/hip_runtime.h>
#include <hip/hip_bf16.h>

// ---------------------------------------------------------------------------
// MultiHeadAttention for MI455X (gfx1250): bf16 WMMA GEMMs + wave32 flash attn
// Round 2: 32-row tiles everywhere -> 2x math per vmem issue slot, half the
// L2 K/V re-read traffic in attention.
// ---------------------------------------------------------------------------

typedef __attribute__((ext_vector_type(16))) __bf16 v16bf;
typedef __attribute__((ext_vector_type(8)))  __bf16 v8bf;
typedef __attribute__((ext_vector_type(8)))  float  v8f;

#define B_SZ    2
#define S_LEN   2048
#define DMODEL  1024
#define NHEADS  16
#define HD      64
#define MASK_VALUE (-1000000.0f)

// A-operand (16x32 bf16, MxK) per-lane load: lane holds K{b..b+15}
// with b = half*16  -> two adjacent 16-byte chunks.
static __device__ __forceinline__ v16bf load_a16(const __bf16* p) {
    v8bf lo = *(const v8bf*)p;
    v8bf hi = *(const v8bf*)(p + 8);
    return __builtin_shufflevector(lo, hi, 0,1,2,3,4,5,6,7,8,9,10,11,12,13,14,15);
}

static __device__ __forceinline__ v8f wmma_bf16(v16bf a, v16bf b, v8f c) {
    // 8 args: (neg_a, A, neg_b, B, c_mod, C, reuse_a, reuse_b)
    return __builtin_amdgcn_wmma_f32_16x16x32_bf16(false, a, false, b,
                                                   (short)0, c, false, false);
}

// ----------------------------- conversions ---------------------------------

__global__ void cvt_f32_bf16_kernel(const float* __restrict__ x,
                                    __bf16* __restrict__ y, int n) {
    int i = blockIdx.x * blockDim.x + threadIdx.x;
    if (i < n) y[i] = (__bf16)x[i];
}

// wt[n*K + k] = w[k*N + n]   (store weights N-major so B-operand is contiguous)
__global__ void transpose_to_bf16_kernel(const float* __restrict__ w,
                                         __bf16* __restrict__ wt,
                                         int Kd, int N) {
    int i = blockIdx.x * blockDim.x + threadIdx.x;
    if (i < Kd * N) {
        int n = i / Kd;
        int k = i - n * Kd;
        wt[i] = (__bf16)w[(size_t)k * N + n];
    }
}

// ------------------------------- GEMM --------------------------------------
// C(M,N) = A(M,K) * Bt(N,K)^T.  One wave per 32x64 tile: each B operand is
// shared by two A row-blocks -> 8 WMMA per 12 b128 loads per K-chunk.
// mode 0: bf16 row-major out;  mode 1: bf16 scatter to Vt[b][h][d][s];
// mode 2: f32 row-major out.
__global__ void gemm_bf16_kernel(const __bf16* __restrict__ A,
                                 const __bf16* __restrict__ Bt,
                                 void* __restrict__ Cout,
                                 int M, int N, int Kd, int mode) {
    const int lane = threadIdx.x;
    const int half = lane >> 4;        // 0: lanes 0-15, 1: lanes 16-31
    const int lrow = lane & 15;
    const int m0 = blockIdx.y * 32;
    const int n0 = blockIdx.x * 64;

    v8f acc[2][4] = {};
    const __bf16* ar0 = A + (size_t)(m0 + lrow) * Kd;
    const __bf16* ar1 = A + (size_t)(m0 + 16 + lrow) * Kd;

    for (int k0 = 0; k0 < Kd; k0 += 32) {
        v16bf a0 = load_a16(ar0 + k0 + half * 16);
        v16bf a1 = load_a16(ar1 + k0 + half * 16);
#pragma unroll
        for (int nt = 0; nt < 4; ++nt) {
            // B-operand: lane column n, 16 contiguous K values at half*16
            const __bf16* bp = Bt + (size_t)(n0 + nt * 16 + lrow) * Kd
                                  + k0 + half * 16;
            v16bf b = *(const v16bf*)bp;
            acc[0][nt] = wmma_bf16(a0, b, acc[0][nt]);
            acc[1][nt] = wmma_bf16(a1, b, acc[1][nt]);
        }
    }

    if (mode == 2) {
        float* C = (float*)Cout;
#pragma unroll
        for (int mt = 0; mt < 2; ++mt)
#pragma unroll
            for (int nt = 0; nt < 4; ++nt)
#pragma unroll
                for (int r = 0; r < 8; ++r)
                    C[(size_t)(m0 + mt * 16 + r + half * 8) * N
                      + n0 + nt * 16 + lrow] = acc[mt][nt][r];
    } else if (mode == 0) {
        __bf16* C = (__bf16*)Cout;
#pragma unroll
        for (int mt = 0; mt < 2; ++mt)
#pragma unroll
            for (int nt = 0; nt < 4; ++nt)
#pragma unroll
                for (int r = 0; r < 8; ++r)
                    C[(size_t)(m0 + mt * 16 + r + half * 8) * N
                      + n0 + nt * 16 + lrow] = (__bf16)acc[mt][nt][r];
    } else {  // mode 1: scatter into Vt[((b*H + h)*HD + d)*S + s]
        __bf16* C = (__bf16*)Cout;
#pragma unroll
        for (int mt = 0; mt < 2; ++mt)
#pragma unroll
            for (int nt = 0; nt < 4; ++nt)
#pragma unroll
                for (int r = 0; r < 8; ++r) {
                    int gr = m0 + mt * 16 + r + half * 8;   // b*S + s
                    int c  = n0 + nt * 16 + lrow;           // h*HD + d
                    int bb = gr >> 11, ss = gr & (S_LEN - 1);
                    int hh = c >> 6,  dd = c & (HD - 1);
                    C[(size_t)((bb * NHEADS + hh) * HD + dd) * S_LEN + ss] =
                        (__bf16)acc[mt][nt][r];
                }
    }
}

// --------------------------- flash attention -------------------------------
// One wave per (b, h, 32-query rows). Online softmax, 32 keys per step.
// K/V operand loads are shared by both query row-blocks: 16 WMMA per 16 global
// b128 loads per key step.
__global__ void flash_attn_kernel(const __bf16* __restrict__ Q,
                                  const __bf16* __restrict__ K,
                                  const __bf16* __restrict__ Vt,
                                  const int* __restrict__ valid_lens,
                                  __bf16* __restrict__ O) {
    __shared__ __bf16 Ptile[32 * 32];   // P re-layout staging (2 KB)

    const int lane = threadIdx.x;
    const int half = lane >> 4;
    const int lrow = lane & 15;
    const int b  = blockIdx.z;
    const int h  = blockIdx.y;
    const int q0 = blockIdx.x * 32;
    const int vl = valid_lens[b];
    const float scale = 0.17677669529663689f;   // 1/sqrt(B*H) = 1/sqrt(32)

    // Q A-operands: two row-blocks x two K-chunks of 32 (hd = 64)
    v16bf qa[2][2];
#pragma unroll
    for (int mt = 0; mt < 2; ++mt) {
        const __bf16* qbase = Q + (size_t)(b * S_LEN + q0 + mt * 16 + lrow) * DMODEL
                                + h * HD;
        qa[mt][0] = load_a16(qbase + half * 16);
        qa[mt][1] = load_a16(qbase + 32 + half * 16);
    }

    v8f acc[2][4] = {};
    float m_i[2][8], l_i[2][8];
#pragma unroll
    for (int mt = 0; mt < 2; ++mt)
#pragma unroll
        for (int r = 0; r < 8; ++r) { m_i[mt][r] = -__builtin_inff(); l_i[mt][r] = 0.0f; }

    for (int j0 = 0; j0 < S_LEN; j0 += 32) {
        // ---- scores: per row-block, two 16x16 tiles (cols j0..+15, +16..+31) ----
        const __bf16* k0p = K + (size_t)(b * S_LEN + j0 + lrow) * DMODEL
                              + h * HD + half * 16;
        const __bf16* k1p = K + (size_t)(b * S_LEN + j0 + 16 + lrow) * DMODEL
                              + h * HD + half * 16;
        v16bf k00 = *(const v16bf*)k0p;
        v16bf k01 = *(const v16bf*)(k0p + 32);
        v16bf k10 = *(const v16bf*)k1p;
        v16bf k11 = *(const v16bf*)(k1p + 32);

        v8f s0[2] = {}, s1[2] = {};
#pragma unroll
        for (int mt = 0; mt < 2; ++mt) {
            s0[mt] = wmma_bf16(qa[mt][0], k00, s0[mt]);
            s0[mt] = wmma_bf16(qa[mt][1], k01, s0[mt]);
            s1[mt] = wmma_bf16(qa[mt][0], k10, s1[mt]);
            s1[mt] = wmma_bf16(qa[mt][1], k11, s1[mt]);
        }

        // ---- scale + sequence mask + online softmax (per row-block) ----
        const int col0 = j0 + lrow;
        const int col1 = col0 + 16;
        float p0[2][8], p1[2][8];
#pragma unroll
        for (int mt = 0; mt < 2; ++mt) {
            float mx[8];
#pragma unroll
            for (int r = 0; r < 8; ++r) {
                float a0 = (col0 < vl) ? s0[mt][r] * scale : MASK_VALUE;
                float a1 = (col1 < vl) ? s1[mt][r] * scale : MASK_VALUE;
                s0[mt][r] = a0; s1[mt][r] = a1;
                mx[r] = fmaxf(a0, a1);
            }
            // row max across the 16-lane half-group (rows mt*16 + r + half*8)
#pragma unroll
            for (int off = 1; off < 16; off <<= 1)
#pragma unroll
                for (int r = 0; r < 8; ++r)
                    mx[r] = fmaxf(mx[r], __shfl_xor(mx[r], off, 32));

            float alpha[8];
#pragma unroll
            for (int r = 0; r < 8; ++r) {
                float mnew = fmaxf(m_i[mt][r], mx[r]);
                alpha[r] = __expf(m_i[mt][r] - mnew);
                p0[mt][r] = __expf(s0[mt][r] - mnew);
                p1[mt][r] = __expf(s1[mt][r] - mnew);
                m_i[mt][r] = mnew;
            }
            float rs[8];
#pragma unroll
            for (int r = 0; r < 8; ++r) rs[r] = p0[mt][r] + p1[mt][r];
#pragma unroll
            for (int off = 1; off < 16; off <<= 1)
#pragma unroll
                for (int r = 0; r < 8; ++r)
                    rs[r] += __shfl_xor(rs[r], off, 32);
#pragma unroll
            for (int r = 0; r < 8; ++r)
                l_i[mt][r] = l_i[mt][r] * alpha[r] + rs[r];

            // rescale running output accumulators
#pragma unroll
            for (int nt = 0; nt < 4; ++nt)
#pragma unroll
                for (int r = 0; r < 8; ++r)
                    acc[mt][nt][r] *= alpha[r];

            // ---- P (16x32, C-layout) -> LDS row-major ----
#pragma unroll
            for (int r = 0; r < 8; ++r) {
                int row = mt * 16 + r + half * 8;
                Ptile[row * 32 + lrow]      = (__bf16)p0[mt][r];
                Ptile[row * 32 + lrow + 16] = (__bf16)p1[mt][r];
            }
        }
        asm volatile("s_wait_dscnt 0" ::: "memory");
        v16bf pa0 = load_a16(&Ptile[(lrow) * 32 + half * 16]);
        v16bf pa1 = load_a16(&Ptile[(16 + lrow) * 32 + half * 16]);

        // ---- acc += P * V  (V stored d-major: Vt[(b*H+h)*HD + d][s]) ----
        const __bf16* vb = Vt + (size_t)((b * NHEADS + h) * HD) * S_LEN
                              + j0 + half * 16;
#pragma unroll
        for (int nt = 0; nt < 4; ++nt) {
            v16bf v = *(const v16bf*)(vb + (size_t)(nt * 16 + lrow) * S_LEN);
            acc[0][nt] = wmma_bf16(pa0, v, acc[0][nt]);
            acc[1][nt] = wmma_bf16(pa1, v, acc[1][nt]);
        }
    }

    // ---- epilogue: normalize by l and store merged-head bf16 ----
#pragma unroll
    for (int mt = 0; mt < 2; ++mt) {
        float inv[8];
#pragma unroll
        for (int r = 0; r < 8; ++r) inv[r] = 1.0f / l_i[mt][r];
#pragma unroll
        for (int nt = 0; nt < 4; ++nt)
#pragma unroll
            for (int r = 0; r < 8; ++r) {
                int row = q0 + mt * 16 + r + half * 8;
                int col = h * HD + nt * 16 + lrow;
                O[(size_t)(b * S_LEN + row) * DMODEL + col] =
                    (__bf16)(acc[mt][nt][r] * inv[r]);
            }
    }
}

// ------------------------------ launcher -----------------------------------

extern "C" void kernel_launch(void* const* d_in, const int* in_sizes, int n_in,
                              void* d_out, int out_size, void* d_ws, size_t ws_size,
                              hipStream_t stream) {
    (void)in_sizes; (void)n_in; (void)out_size; (void)ws_size;

    const float* q    = (const float*)d_in[0];
    const float* k    = (const float*)d_in[1];
    const float* v    = (const float*)d_in[2];
    const int*   vlen = (const int*)d_in[3];
    const float* Wq   = (const float*)d_in[4];
    const float* Wk   = (const float*)d_in[5];
    const float* Wv   = (const float*)d_in[6];
    const float* Wo   = (const float*)d_in[7];
    float* out = (float*)d_out;

    constexpr int ROWS = B_SZ * S_LEN;          // 4096
    constexpr size_t XB = (size_t)ROWS * DMODEL * sizeof(__bf16);   // 8 MB
    constexpr size_t WB = (size_t)DMODEL * DMODEL * sizeof(__bf16); // 2 MB

    char* ws = (char*)d_ws;
    size_t off = 0;
    auto take = [&](size_t bytes) -> char* {
        char* p = ws + off;
        off += (bytes + 255) & ~(size_t)255;
        return p;
    };
    __bf16* Xq  = (__bf16*)take(XB);
    __bf16* Xk  = (__bf16*)take(XB);
    __bf16* Xv  = (__bf16*)take(XB);
    __bf16* Wqt = (__bf16*)take(WB);
    __bf16* Wkt = (__bf16*)take(WB);
    __bf16* Wvt = (__bf16*)take(WB);
    __bf16* Wot = (__bf16*)take(WB);
    __bf16* Qp  = (__bf16*)take(XB);
    __bf16* Kp  = (__bf16*)take(XB);
    __bf16* Vtr = (__bf16*)take(XB);
    __bf16* Ob  = (__bf16*)take(XB);

    const int nX = ROWS * DMODEL;
    const int nW = DMODEL * DMODEL;
    cvt_f32_bf16_kernel<<<(nX + 255) / 256, 256, 0, stream>>>(q, Xq, nX);
    cvt_f32_bf16_kernel<<<(nX + 255) / 256, 256, 0, stream>>>(k, Xk, nX);
    cvt_f32_bf16_kernel<<<(nX + 255) / 256, 256, 0, stream>>>(v, Xv, nX);
    transpose_to_bf16_kernel<<<(nW + 255) / 256, 256, 0, stream>>>(Wq, Wqt, DMODEL, DMODEL);
    transpose_to_bf16_kernel<<<(nW + 255) / 256, 256, 0, stream>>>(Wk, Wkt, DMODEL, DMODEL);
    transpose_to_bf16_kernel<<<(nW + 255) / 256, 256, 0, stream>>>(Wv, Wvt, DMODEL, DMODEL);
    transpose_to_bf16_kernel<<<(nW + 255) / 256, 256, 0, stream>>>(Wo, Wot, DMODEL, DMODEL);

    dim3 gg(DMODEL / 64, ROWS / 32);            // (16, 128) 32x64 tiles, 1 wave each
    gemm_bf16_kernel<<<gg, 32, 0, stream>>>(Xq, Wqt, Qp,  ROWS, DMODEL, DMODEL, 0);
    gemm_bf16_kernel<<<gg, 32, 0, stream>>>(Xk, Wkt, Kp,  ROWS, DMODEL, DMODEL, 0);
    gemm_bf16_kernel<<<gg, 32, 0, stream>>>(Xv, Wvt, Vtr, ROWS, DMODEL, DMODEL, 1);

    dim3 ga(S_LEN / 32, NHEADS, B_SZ);          // (64, 16, 2)
    flash_attn_kernel<<<ga, 32, 0, stream>>>(Qp, Kp, Vtr, vlen, Ob);

    gemm_bf16_kernel<<<gg, 32, 0, stream>>>(Ob, Wot, out, ROWS, DMODEL, DMODEL, 2);
}
